// CrossAttentionWithTemporalEmbedding_67053029425312
// MI455X (gfx1250) — matrix operand order repaired and checked by reading, not compiled, not run on hardware
//
#include <hip/hip_runtime.h>

// Cross-attention w/ temporal embedding, MI455X (gfx1250), wave32 + WMMA f16.
// Pass 1: fuse (f + t_emb) -> q/k/v projections via v_wmma_f32_16x16x32_f16,
//         store q,k as f16 [b][n][32] and v as f16 [b][c][n].
// Pass 2: flash attention (online softmax) with WMMA for S = Q K^T and O += P V,
//         K/V chunk staging via GLOBAL_LOAD_ASYNC_TO_LDS (ASYNCcnt-tracked).

constexpr int CH  = 256;   // channels
constexpr int QCH = 32;    // q/k channels
constexpr int NB  = 4;     // batch
constexpr int HWS = 4096;  // spatial (64*64)

typedef __attribute__((ext_vector_type(16))) _Float16 v16h;
typedef __attribute__((ext_vector_type(8)))  _Float16 v8h;
typedef __attribute__((ext_vector_type(4)))  _Float16 v4h;
typedef __attribute__((ext_vector_type(8)))  float    v8f;
typedef __attribute__((ext_vector_type(4)))  float    v4f;

// async-to-LDS builtin operand types (from compiler diagnostics):
typedef int v2i __attribute__((vector_size(8)));
typedef int v4i __attribute__((vector_size(16)));

#define AS1 __attribute__((address_space(1)))
#define AS3 __attribute__((address_space(3)))

#if __has_builtin(__builtin_amdgcn_global_load_async_to_lds_b128) && \
    __has_builtin(__builtin_amdgcn_global_load_async_to_lds_b64)
#define HAVE_ASYNC_LDS 1
#else
#define HAVE_ASYNC_LDS 0
#endif

#if __has_builtin(__builtin_amdgcn_s_wait_asynccnt)
#define WAIT_ASYNC0() __builtin_amdgcn_s_wait_asynccnt(0)
#else
#define WAIT_ASYNC0() asm volatile("s_wait_asynccnt 0x0" ::: "memory")
#endif

__device__ __forceinline__ v16h pack16(v8h lo, v8h hi) {
  v16h r;
#pragma unroll
  for (int i = 0; i < 8; ++i) { r[i] = lo[i]; r[i + 8] = hi[i]; }
  return r;
}

// B-operand (32x16 f16): lane holds N=lane%16, elements i=0..15 are K=g*16+i,
// i.e. 16 contiguous halfs starting at p.
__device__ __forceinline__ v16h loadB16(const _Float16* p) {
  v8h lo = *(const v8h*)p;
  v8h hi = *(const v8h*)(p + 8);
  return pack16(lo, hi);
}

// A-operand (16x32 f16): lane m=lane%16, halfs i=0..15 -> K=g*16+i.
__device__ __forceinline__ v16h loadA16(const _Float16* row, int g) {
  v8h lo = *(const v8h*)(row + g * 16);
  v8h hi = *(const v8h*)(row + g * 16 + 8);
  return pack16(lo, hi);
}

// reductions across each 16-lane half (rows are replicated per half in C/D layout)
__device__ __forceinline__ float hmax16(float x) {
#pragma unroll
  for (int off = 8; off >= 1; off >>= 1) x = fmaxf(x, __shfl_xor(x, off, 32));
  return x;
}
__device__ __forceinline__ float hsum16(float x) {
#pragma unroll
  for (int off = 8; off >= 1; off >>= 1) x += __shfl_xor(x, off, 32);
  return x;
}

// ---------------------------------------------------------------------------
// Pass 1: projections. grid = (HWS/64, NB, 2 streams), block = 128 (4 waves).
// ---------------------------------------------------------------------------
__global__ __launch_bounds__(128)
void proj_kernel(const float* __restrict__ f1, const float* __restrict__ f2,
                 const float* __restrict__ te1, const float* __restrict__ te2,
                 const float* __restrict__ wq, const float* __restrict__ bq,
                 const float* __restrict__ wk, const float* __restrict__ bk,
                 const float* __restrict__ wv, const float* __restrict__ bv,
                 _Float16* __restrict__ qg, _Float16* __restrict__ kg,
                 _Float16* __restrict__ vg) {
  __shared__ _Float16 fe[64][CH + 8];  // [spatial row][channel], f16, ~33.8 KB

  const int s = blockIdx.z;
  const int b = blockIdx.y;
  const int nblock = blockIdx.x * 64;
  const float* f  = s ? f2  : f1;
  const float* te = s ? te2 : te1;
  const size_t QN = (size_t)NB * HWS * QCH;
  const size_t VN = (size_t)NB * CH * HWS;
  qg += (size_t)s * QN;
  kg += (size_t)s * QN;
  vg += (size_t)s * VN;

  const int tid  = threadIdx.x;
  const int lane = tid & 31;
  const int wv32 = tid >> 5;        // wave 0..3
  const int g    = lane >> 4;       // half-wave
  const int l16  = lane & 15;

  // Stage fe = f + t_emb, transposed to [row][ch] f16. 4096 float4 slots.
#pragma unroll 4
  for (int it = 0; it < 32; ++it) {
    int slot = it * 128 + tid;
    int c    = slot >> 4;           // 0..255
    int col4 = (slot & 15) * 4;     // 0..60
    v4f x = *(const v4f*)(f + (size_t)b * CH * HWS + (size_t)c * HWS + nblock + col4);
    float t = te[c];
#pragma unroll
    for (int j = 0; j < 4; ++j) fe[col4 + j][c] = (_Float16)(x[j] + t);
  }
  __syncthreads();

  const int rowbase = wv32 * 16;    // local query-row base for this wave

  for (int tile = 0; tile < 20; ++tile) {
    const float* wmat; const float* bias; int obase;
    if (tile < 2)      { wmat = wq; bias = bq; obase = tile * 16; }
    else if (tile < 4) { wmat = wk; bias = bk; obase = (tile - 2) * 16; }
    else               { wmat = wv; bias = bv; obase = (tile - 4) * 16; }
    const int ocol = obase + l16;

    v8f acc = {};
#pragma unroll 2
    for (int ks = 0; ks < CH; ks += 32) {
      v16h A = loadA16(&fe[rowbase + l16][ks], g);
      // B[k=c][n=o] = w[o][c]: 16 contiguous fp32, convert to f16
      const float* wrow = wmat + (size_t)ocol * CH + ks + g * 16;
      v4f w0 = *(const v4f*)(wrow + 0);
      v4f w1 = *(const v4f*)(wrow + 4);
      v4f w2 = *(const v4f*)(wrow + 8);
      v4f w3 = *(const v4f*)(wrow + 12);
      v16h B;
#pragma unroll
      for (int i = 0; i < 4; ++i) {
        B[i]      = (_Float16)w0[i];
        B[4 + i]  = (_Float16)w1[i];
        B[8 + i]  = (_Float16)w2[i];
        B[12 + i] = (_Float16)w3[i];
      }
      acc = __builtin_amdgcn_wmma_f32_16x16x32_f16(false, A, false, B,
                                                   (short)0, acc, false, false);
    }
    const float bb = bias[ocol];
    if (tile < 4) {
      // q/k layout [b][n][QCH]; D row M = r + 8g -> n, col N = l16 -> o
      _Float16* outp = (tile < 2) ? qg : kg;
#pragma unroll
      for (int r = 0; r < 8; ++r) {
        int n = nblock + rowbase + r + 8 * g;
        outp[((size_t)b * HWS + n) * QCH + ocol] = (_Float16)(acc[r] + bb);
      }
    } else {
      // v layout [b][c][n]; rows r are 8 consecutive n -> one b128 store
      int n0 = nblock + rowbase + 8 * g;
      v8h pk;
#pragma unroll
      for (int r = 0; r < 8; ++r) pk[r] = (_Float16)(acc[r] + bb);
      *(v8h*)&vg[((size_t)b * CH + ocol) * HWS + n0] = pk;
    }
  }
}

// ---------------------------------------------------------------------------
// Pass 2: flash attention. grid = (HWS/128, NB, 2), block = 256 (8 waves).
// stream 0: out1 = gamma * softmax(Q2 K1^T) V1 + f1
// stream 1: out2 = gamma * softmax(Q1 K2^T) V2 + f2
// ---------------------------------------------------------------------------
__global__ __launch_bounds__(256)
void attn_kernel(const float* __restrict__ f1, const float* __restrict__ f2,
                 const float* __restrict__ gamma,
                 const _Float16* __restrict__ qg, const _Float16* __restrict__ kg,
                 const _Float16* __restrict__ vg, float* __restrict__ out) {
  __shared__ _Float16 kl[32][40];        // key chunk [key][ch], 2.5 KB
  __shared__ _Float16 vl[CH][40];        // value chunk [c][key], 20 KB
  __shared__ _Float16 pl[8][16 * 32];    // per-wave P tile [m][key], 8 KB

  const int s = blockIdx.z;
  const int b = blockIdx.y;
  const int qbase = blockIdx.x * 128;
  const size_t QN = (size_t)NB * HWS * QCH;
  const size_t VN = (size_t)NB * CH * HWS;

  const _Float16* q  = qg + (s ? 0 : QN) + (size_t)b * HWS * QCH;  // s0: q2
  const _Float16* kk = kg + (s ? QN : 0) + (size_t)b * HWS * QCH;  // s0: k1
  const _Float16* vv = vg + (s ? VN : 0) + (size_t)b * CH * HWS;   // s0: v1
  const float* fres  = (s ? f2 : f1) + (size_t)b * CH * HWS;
  float* o = out + (size_t)s * NB * CH * HWS + (size_t)b * CH * HWS;

  const int tid  = threadIdx.x;
  const int lane = tid & 31;
  const int wv32 = tid >> 5;        // wave 0..7
  const int g    = lane >> 4;
  const int l16  = lane & 15;
  const float gm = gamma[0];

  // Per-thread staging addresses (constant across chunks).
  const int skey  = tid >> 3;           // K staging: key row, 8 threads/row
  const int scoff = (tid & 7) * 4;      // K staging: 4 halfs each
  const _Float16* kk_src = &kk[(size_t)skey * QCH + scoff];
  const _Float16* vv_src = &vv[(size_t)tid * HWS];

  // Q A-operand: one 16x32 tile per wave, loaded once.
  const _Float16* qrow = q + (size_t)(qbase + wv32 * 16 + l16) * QCH;
  v16h Aq = loadA16(qrow, g);

  v8f acc[16];
#pragma unroll
  for (int t = 0; t < 16; ++t) acc[t] = (v8f){};
  float mrow[8], lrow[8];
#pragma unroll
  for (int r = 0; r < 8; ++r) { mrow[r] = -3.0e38f; lrow[r] = 0.0f; }

  for (int kb = 0; kb < HWS; kb += 32) {
    __syncthreads();
#if HAVE_ASYNC_LDS
    // Async DMA of K (b64/thread) and V (4 x b128/thread) chunks into LDS;
    // tracked with ASYNCcnt, keeps staging off the VALU/VGPR path.
    __builtin_amdgcn_global_load_async_to_lds_b64(
        (AS1 v2i*)(kk_src + (size_t)kb * QCH), (AS3 v2i*)&kl[skey][scoff], 0, 0);
    {
      AS1 v4i* gsrc = (AS1 v4i*)(vv_src + kb);
      AS3 v4i* ldst = (AS3 v4i*)&vl[tid][0];
      __builtin_amdgcn_global_load_async_to_lds_b128(gsrc, ldst, 0, 0);
      __builtin_amdgcn_global_load_async_to_lds_b128(gsrc, ldst, 16, 0);
      __builtin_amdgcn_global_load_async_to_lds_b128(gsrc, ldst, 32, 0);
      __builtin_amdgcn_global_load_async_to_lds_b128(gsrc, ldst, 48, 0);
    }
    WAIT_ASYNC0();
#else
    // Stage K chunk: 32 keys x 32 ch (8 threads per key row, 4 halfs each)
    *(v4h*)&kl[skey][scoff] = *(const v4h*)(kk_src + (size_t)kb * QCH);
    // Stage V chunk: one channel row per thread, 32 contiguous halfs
    {
      const _Float16* src = vv_src + kb;
      *(v8h*)&vl[tid][0]  = *(const v8h*)(src + 0);
      *(v8h*)&vl[tid][8]  = *(const v8h*)(src + 8);
      *(v8h*)&vl[tid][16] = *(const v8h*)(src + 16);
      *(v8h*)&vl[tid][24] = *(const v8h*)(src + 24);
    }
#endif
    if (kb + 32 < HWS) {  // uniform; prefetch next chunk into cache
      __builtin_prefetch(vv_src + kb + 32, 0, 1);
      __builtin_prefetch(kk_src + (size_t)(kb + 32) * QCH, 0, 1);
    }
    __syncthreads();

    // S = Q K^T for the two 16-key sub-tiles (single K=32 WMMA each)
    v16h B0 = loadB16(&kl[l16][g * 16]);
    v16h B1 = loadB16(&kl[16 + l16][g * 16]);
    v8f s0 = {}, s1 = {};
    s0 = __builtin_amdgcn_wmma_f32_16x16x32_f16(false, Aq, false, B0, (short)0, s0, false, false);
    s1 = __builtin_amdgcn_wmma_f32_16x16x32_f16(false, Aq, false, B1, (short)0, s1, false, false);

    // Online softmax; write P (f16) to per-wave LDS for A-layout re-shape.
    float scale[8];
    bool changed = false;
#pragma unroll
    for (int r = 0; r < 8; ++r) {
      float x0 = s0[r], x1 = s1[r];
      float mx = hmax16(fmaxf(x0, x1));
      float mo = mrow[r];
      float mn = fmaxf(mo, mx);
      changed = changed || (mn > mo);
      float sc = __expf(mo - mn);
      float p0 = __expf(x0 - mn);
      float p1 = __expf(x1 - mn);
      float rs = hsum16(p0 + p1);
      lrow[r] = lrow[r] * sc + rs;
      mrow[r] = mn;
      scale[r] = sc;
      int row = r + 8 * g;
      pl[wv32][row * 32 + l16]      = (_Float16)p0;
      pl[wv32][row * 32 + 16 + l16] = (_Float16)p1;
    }
    // Rescale accumulators only when some row-max moved (wave-uniform branch;
    // steady-state online softmax almost never rescales -> saves 128 v_mul/chunk).
    if (__ballot(changed)) {
#pragma unroll
      for (int t = 0; t < 16; ++t)
#pragma unroll
        for (int r = 0; r < 8; ++r) acc[t][r] *= scale[r];
    }

    // O += P V : A = P (16x32), 16 B-tiles across C=256
    v16h Ap = loadA16(&pl[wv32][l16 * 32], g);
#pragma unroll
    for (int t = 0; t < 16; ++t) {
      v16h Bv = loadB16(&vl[t * 16 + l16][g * 16]);
      acc[t] = __builtin_amdgcn_wmma_f32_16x16x32_f16(false, Ap, false, Bv,
                                                      (short)0, acc[t], false, false);
    }
  }

  // Epilogue: out[c][n] = gamma * acc/l + f ; rows r are consecutive n.
  const int n0 = qbase + wv32 * 16 + 8 * g;
  float inv[8];
#pragma unroll
  for (int r = 0; r < 8; ++r) inv[r] = 1.0f / lrow[r];
#pragma unroll
  for (int t = 0; t < 16; ++t) {
    int c = t * 16 + l16;
    const float* fp = fres + (size_t)c * HWS + n0;
    float* op = o + (size_t)c * HWS + n0;
    v4f fa = *(const v4f*)(fp);
    v4f fb = *(const v4f*)(fp + 4);
    v4f oa, ob;
#pragma unroll
    for (int r = 0; r < 4; ++r) {
      oa[r] = gm * acc[t][r] * inv[r] + fa[r];
      ob[r] = gm * acc[t][r + 4] * inv[r + 4] + fb[r];
    }
    *(v4f*)(op)     = oa;
    *(v4f*)(op + 4) = ob;
  }
}

// ---------------------------------------------------------------------------
extern "C" void kernel_launch(void* const* d_in, const int* in_sizes, int n_in,
                              void* d_out, int out_size, void* d_ws, size_t ws_size,
                              hipStream_t stream) {
  (void)in_sizes; (void)n_in; (void)out_size; (void)ws_size;
  const float* f1  = (const float*)d_in[0];
  const float* f2  = (const float*)d_in[1];
  const float* te1 = (const float*)d_in[2];
  const float* te2 = (const float*)d_in[3];
  const float* wq  = (const float*)d_in[4];
  const float* bq  = (const float*)d_in[5];
  const float* wk  = (const float*)d_in[6];
  const float* bk  = (const float*)d_in[7];
  const float* wv  = (const float*)d_in[8];
  const float* bv  = (const float*)d_in[9];
  const float* gamma = (const float*)d_in[10];

  _Float16* ws = (_Float16*)d_ws;
  const size_t QN = (size_t)NB * HWS * QCH;   // 524288 halfs per q/k tensor
  const size_t VN = (size_t)NB * CH * HWS;    // 4194304 halfs per v tensor
  _Float16* qg = ws;                 // q1, q2
  _Float16* kg = ws + 2 * QN;        // k1, k2
  _Float16* vg = ws + 4 * QN;        // v1, v2   (total 20 MB)

  dim3 gp(HWS / 64, NB, 2);
  proj_kernel<<<gp, 128, 0, stream>>>(f1, f2, te1, te2, wq, bq, wk, bk, wv, bv,
                                      qg, kg, vg);
  dim3 ga(HWS / 128, NB, 2);
  attn_kernel<<<ga, 256, 0, stream>>>(f1, f2, gamma, qg, kg, vg, (float*)d_out);
}
